// MOE_88081189306698
// MI455X (gfx1250) — matrix operand order repaired and checked by reading, not compiled, not run on hardware
//
#include <hip/hip_runtime.h>

// ---------------------------------------------------------------------------
// MoE top-2 FFN for MI455X (gfx1250), wave32 + WMMA bf16 + async global->LDS.
//
// Pipeline (all on `stream`):
//   0) zero y + expert counts
//   1) convert x -> bf16 (same layout)
//   2) transpose+convert W1[e] (D,H)->(H,D) bf16 ; W2[e] (H,D)->(D,H) bf16
//      (so WMMA B-fragments read K-contiguous 16B runs per lane)
//   3) router: logits=x@Wg, softmax, top-2, renorm weights, per-expert counts
//   4) plan:   prefix offsets + tile descriptors (1 thread, E=8)
//   5) scatter: deterministic in-order grouping of (token,weight) by expert
//   6) ffn:    x tile async-copied to LDS (GLOBAL_LOAD_ASYNC_TO_LDS_B128 +
//              s_wait_asynccnt), grouped GEMM1 -> b1+ReLU -> (LDS relayout)
//              -> GEMM2 -> b2, scale by combine weight, atomic-add into y
// ---------------------------------------------------------------------------

#define N_TOK 16384   // B*S
#define D_    1024
#define E_    8
#define H_    2048
#define TM    16
#define MAXTILES ((N_TOK * 2) / TM + E_)   // 2056
#define XROWB 2064   // 1032 bf16: stride/4 % 64 == 4 -> conflict-free A loads
#define HROWB 272    // 136 bf16:  stride/4 % 64 == 4

typedef __bf16 bf16_t;
typedef bf16_t v16bf __attribute__((ext_vector_type(16)));
typedef float  v8f   __attribute__((ext_vector_type(8)));
typedef unsigned int u32x4 __attribute__((ext_vector_type(4)));
typedef float f32x4 __attribute__((ext_vector_type(4)));

union Frag { v16bf v; u32x4 q[2]; };

__device__ __forceinline__ unsigned short f2bf(float f) {
  unsigned u = __float_as_uint(f);
  u += 0x7fffu + ((u >> 16) & 1u);          // round-to-nearest-even
  return (unsigned short)(u >> 16);
}

// ---------------------------------------------------------------- kernels ---

__global__ void zero_init_kernel(float* __restrict__ y, int* __restrict__ counts, long n4) {
  long i = (long)blockIdx.x * blockDim.x + threadIdx.x;
  if (i < n4) ((f32x4*)y)[i] = (f32x4){0.f, 0.f, 0.f, 0.f};
  if (blockIdx.x == 0 && threadIdx.x < E_) counts[threadIdx.x] = 0;
}

__global__ void cvt_bf16_kernel(const float* __restrict__ s, unsigned short* __restrict__ d, long n8) {
  long i = (long)blockIdx.x * blockDim.x + threadIdx.x;
  if (i >= n8) return;
  const f32x4* sp = (const f32x4*)s + i * 2;
  f32x4 a = sp[0], b = sp[1];
  union { u32x4 q; unsigned short h[8]; } o;
  o.h[0] = f2bf(a[0]); o.h[1] = f2bf(a[1]); o.h[2] = f2bf(a[2]); o.h[3] = f2bf(a[3]);
  o.h[4] = f2bf(b[0]); o.h[5] = f2bf(b[1]); o.h[6] = f2bf(b[2]); o.h[7] = f2bf(b[3]);
  ((u32x4*)d)[i] = o.q;
}

// src: [E][R][C] f32  ->  dst: [E][C][R] bf16    (block 32x8, tile 32x32)
__global__ void transpose_cvt_kernel(const float* __restrict__ src,
                                     unsigned short* __restrict__ dst, int R, int C) {
  __shared__ unsigned short tile[32][33];
  long e = blockIdx.z;
  const float* s = src + e * (long)R * C;
  unsigned short* d = dst + e * (long)R * C;
  int c0 = blockIdx.x * 32, r0 = blockIdx.y * 32;
  int tx = threadIdx.x, ty = threadIdx.y;
  #pragma unroll
  for (int i = 0; i < 32; i += 8)
    tile[ty + i][tx] = f2bf(s[(long)(r0 + ty + i) * C + (c0 + tx)]);
  __syncthreads();
  #pragma unroll
  for (int i = 0; i < 32; i += 8)
    d[(long)(c0 + ty + i) * R + (r0 + tx)] = tile[tx][ty + i];
}

// one wave per token; 8 tokens per 256-thread block
__global__ void router_kernel(const float* __restrict__ x, const float* __restrict__ Wg,
                              int* __restrict__ tki, float* __restrict__ tkw,
                              int* __restrict__ counts) {
  int wave = threadIdx.x >> 5, lane = threadIdx.x & 31;
  int t = blockIdx.x * 8 + wave;
  const float* xr = x + (long)t * D_;
  float acc[E_];
  #pragma unroll
  for (int e = 0; e < E_; e++) acc[e] = 0.f;
  for (int i = 0; i < 8; i++) {
    int dd = (i * 32 + lane) * 4;
    f32x4 xv = *(const f32x4*)(xr + dd);
    #pragma unroll
    for (int j = 0; j < 4; j++) {
      float xs = xv[j];
      const f32x4* wr = (const f32x4*)(Wg + (long)(dd + j) * E_);
      f32x4 w0 = wr[0], w1 = wr[1];
      acc[0] += xs * w0[0]; acc[1] += xs * w0[1]; acc[2] += xs * w0[2]; acc[3] += xs * w0[3];
      acc[4] += xs * w1[0]; acc[5] += xs * w1[1]; acc[6] += xs * w1[2]; acc[7] += xs * w1[3];
    }
  }
  #pragma unroll
  for (int off = 16; off > 0; off >>= 1) {
    #pragma unroll
    for (int e = 0; e < E_; e++) acc[e] += __shfl_xor(acc[e], off, 32);
  }
  if (lane == 0) {
    float mx = acc[0];
    #pragma unroll
    for (int e = 1; e < E_; e++) mx = fmaxf(mx, acc[e]);
    float p[E_]; float Z = 0.f;
    #pragma unroll
    for (int e = 0; e < E_; e++) { p[e] = __expf(acc[e] - mx); Z += p[e]; }
    int i0 = 0;
    #pragma unroll
    for (int e = 1; e < E_; e++) if (p[e] > p[i0]) i0 = e;     // ties -> lowest idx
    int i1 = (i0 == 0) ? 1 : 0;
    #pragma unroll
    for (int e = 0; e < E_; e++) if (e != i0 && p[e] > p[i1]) i1 = e;
    float p0 = p[i0] / Z, p1 = p[i1] / Z;
    float s = p0 + p1 + 1e-9f;
    tki[t * 2] = i0; tki[t * 2 + 1] = i1;
    tkw[t * 2] = p0 / s; tkw[t * 2 + 1] = p1 / s;
    atomicAdd(&counts[i0], 1);
    atomicAdd(&counts[i1], 1);
  }
}

__global__ void plan_kernel(const int* __restrict__ counts, int* __restrict__ offsets,
                            int* __restrict__ tE, int* __restrict__ tS, int* __restrict__ nT) {
  int off = 0, T = 0;
  for (int e = 0; e < E_; e++) {
    offsets[e] = off;
    int c = counts[e];
    for (int t = 0; t < c; t += TM) {
      if (T < MAXTILES) { tE[T] = e; tS[T] = off + t; T++; }
    }
    off += c;
  }
  nT[0] = T;
}

// deterministic in-order grouping: one wave per expert scans tokens in order
__global__ void scatter_kernel(const int* __restrict__ tki, const float* __restrict__ tkw,
                               const int* __restrict__ offsets,
                               int* __restrict__ etok, float* __restrict__ ew) {
  int e = blockIdx.x, lane = threadIdx.x;
  int base = offsets[e];
  for (int t0 = 0; t0 < N_TOK; t0 += 32) {
    int t = t0 + lane;
    int i0 = tki[t * 2], i1 = tki[t * 2 + 1];
    bool m0 = (i0 == e), m1 = (i1 == e);
    unsigned mask = (unsigned)__ballot(m0 || m1);      // wave32: low 32 bits
    if (m0 || m1) {
      int pos = base + __popc(mask & ((1u << lane) - 1u));
      etok[pos] = t;
      ew[pos] = m0 ? tkw[t * 2] : tkw[t * 2 + 1];
    }
    base += __popc(mask);
  }
}

// ----------------------------------------------------------- FFN tile kernel
__global__ __launch_bounds__(256) void moe_ffn_kernel(
    const unsigned short* __restrict__ xb,
    const unsigned short* __restrict__ w1t,   // [E][H][D] bf16 (transposed)
    const unsigned short* __restrict__ w2t,   // [E][D][H] bf16 (transposed)
    const float* __restrict__ b1g, const float* __restrict__ b2g,
    const int* __restrict__ etok, const float* __restrict__ ew,
    const int* __restrict__ counts, const int* __restrict__ offsets,
    const int* __restrict__ tE, const int* __restrict__ tS, const int* __restrict__ nT,
    float* __restrict__ y) {
  if ((int)blockIdx.x >= nT[0]) return;      // block-uniform: EXEC stays all-1s

  int e = tE[blockIdx.x];
  int start = tS[blockIdx.x];
  int valid = offsets[e] + counts[e] - start;
  if (valid > TM) valid = TM;

  __shared__ __align__(16) unsigned char xsm[TM * XROWB];
  __shared__ __align__(16) unsigned char hsm[TM * HROWB];
  __shared__ int   toks[TM];
  __shared__ float wts[TM];

  int tid = threadIdx.x;
  if (tid < TM) {
    int tk = (tid < valid) ? etok[start + tid] : -1;
    toks[tid] = tk;
    wts[tid] = (tid < valid) ? ew[start + tid] : 0.f;
  }
  __syncthreads();

  { // gather 16 token rows of bf16 x straight into LDS via async DMA.
    // Rows m >= valid alias token 0: garbage there never reaches valid output
    // rows (GEMMs are row-independent; epilogue skips m >= valid).
    int row = tid >> 4, c = tid & 15;
    int tk = toks[row];
    if (tk < 0) tk = 0;
    const unsigned short* gp = xb + (long)tk * D_ + c * 64;           // 128 B
    unsigned lds = (unsigned)(unsigned long long)(xsm + row * XROWB + c * 128);
    #pragma unroll
    for (int q = 0; q < 8; q++) {    // 8 x 16B = 128 B per thread
      asm volatile("global_load_async_to_lds_b128 %0, %1, off"
                   :: "v"(lds), "v"(gp) : "memory");
      lds += 16;
      gp += 8;
    }
    asm volatile("s_wait_asynccnt 0x0" ::: "memory");
  }
  __syncthreads();

  int wid = tid >> 5, lane = tid & 31;
  int n = lane & 15, half = lane >> 4;     // C/B fragment column, K-half
  int m = lane & 15;                       // A fragment row
  int koff = half * 8;                     // B: lanes 16-31 hold K=8..15,24..31
  const unsigned char* axrow = xsm + m * XROWB + half * 16;  // A: K = half*8 ...
  const unsigned char* ahrow = hsm + m * HROWB + half * 16;
  const unsigned short* w1e = w1t + (long)e * H_ * D_;
  const unsigned short* w2e = w2t + (long)e * D_ * H_;

  v8f oacc[8];
  #pragma unroll
  for (int t = 0; t < 8; t++) oacc[t] = (v8f){0.f,0.f,0.f,0.f,0.f,0.f,0.f,0.f};

  for (int hc = 0; hc < H_; hc += 128) {
    // warm L2 for next chunk's B panels (global_prefetch_b8)
    if (hc + 128 < H_) {
      __builtin_prefetch(w1e + (long)(hc + 128 + wid * 16 + n) * D_ + koff, 0, 1);
      __builtin_prefetch(w2e + (long)(wid * 128 + n) * H_ + hc + 128 + koff, 0, 1);
    }

    // ---- GEMM1: this wave's 16x16 h-tile at columns [hc + wid*16, +16)
    int hcol = hc + wid * 16;
    v8f hacc = (v8f){0.f,0.f,0.f,0.f,0.f,0.f,0.f,0.f};
    const unsigned short* bp = w1e + (long)(hcol + n) * D_ + koff;
    #pragma unroll 4
    for (int k0 = 0; k0 < D_; k0 += 32) {
      Frag A, Bf;
      A.q[0] = *(const u32x4*)(axrow + k0 * 2);
      A.q[1] = *(const u32x4*)(axrow + k0 * 2 + 32);
      Bf.q[0] = *(const u32x4*)(bp + k0);
      Bf.q[1] = *(const u32x4*)(bp + k0 + 16);
      hacc = __builtin_amdgcn_wmma_f32_16x16x32_bf16(false, A.v, false, Bf.v,
                                                     (short)0, hacc, false, false);
    }
    float b1v = b1g[e * H_ + hcol + n];

    __syncthreads();   // previous chunk's GEMM2 reads of hsm are done
    #pragma unroll
    for (int i = 0; i < 8; i++) {               // C layout: row = i + half*8
      float hv = fmaxf(hacc[i] + b1v, 0.f);
      *(unsigned short*)(hsm + (i + half * 8) * HROWB + (wid * 16 + n) * 2) = f2bf(hv);
    }
    __syncthreads();   // h chunk visible to all waves

    // ---- GEMM2: accumulate h[16,128] @ W2[hc:hc+128, :] into this wave's cols
    #pragma unroll
    for (int t = 0; t < 8; t++) {
      int col = wid * 128 + t * 16 + n;
      const unsigned short* b2p = w2e + (long)col * H_ + hc + koff;
      #pragma unroll
      for (int k0 = 0; k0 < 128; k0 += 32) {
        Frag A, Bf;
        A.q[0] = *(const u32x4*)(ahrow + k0 * 2);
        A.q[1] = *(const u32x4*)(ahrow + k0 * 2 + 32);
        Bf.q[0] = *(const u32x4*)(b2p + k0);
        Bf.q[1] = *(const u32x4*)(b2p + k0 + 16);
        oacc[t] = __builtin_amdgcn_wmma_f32_16x16x32_bf16(false, A.v, false, Bf.v,
                                                          (short)0, oacc[t], false, false);
      }
    }
  }

  // ---- epilogue: y[token] += w_token * (out + b2[e])
  #pragma unroll
  for (int t = 0; t < 8; t++) {
    int col = wid * 128 + t * 16 + n;
    float b2v = b2g[e * D_ + col];
    #pragma unroll
    for (int i = 0; i < 8; i++) {
      int rowm = i + half * 8;
      if (rowm < valid) {
        float v = wts[rowm] * (oacc[t][i] + b2v);
        unsafeAtomicAdd(&y[(long)toks[rowm] * D_ + col], v);  // global_atomic_add_f32
      }
    }
  }
}

// ----------------------------------------------------------------- launch ---

static inline size_t align256(size_t v) { return (v + 255) & ~(size_t)255; }

extern "C" void kernel_launch(void* const* d_in, const int* in_sizes, int n_in,
                              void* d_out, int out_size, void* d_ws, size_t ws_size,
                              hipStream_t stream) {
  (void)in_sizes; (void)n_in; (void)out_size; (void)ws_size;
  const float* x  = (const float*)d_in[0];
  const float* Wg = (const float*)d_in[1];
  const float* W1 = (const float*)d_in[2];
  const float* b1 = (const float*)d_in[3];
  const float* W2 = (const float*)d_in[4];
  const float* b2 = (const float*)d_in[5];
  float* y = (float*)d_out;

  char* ws = (char*)d_ws;
  size_t off = 0;
  unsigned short* xb  = (unsigned short*)(ws + off); off = align256(off + (size_t)N_TOK * D_ * 2);
  unsigned short* w1t = (unsigned short*)(ws + off); off = align256(off + (size_t)E_ * D_ * H_ * 2);
  unsigned short* w2t = (unsigned short*)(ws + off); off = align256(off + (size_t)E_ * H_ * D_ * 2);
  int*   tki   = (int*)(ws + off);   off = align256(off + (size_t)N_TOK * 2 * 4);
  float* tkw   = (float*)(ws + off); off = align256(off + (size_t)N_TOK * 2 * 4);
  int*   etok  = (int*)(ws + off);   off = align256(off + (size_t)N_TOK * 2 * 4);
  float* ew    = (float*)(ws + off); off = align256(off + (size_t)N_TOK * 2 * 4);
  int*   counts  = (int*)(ws + off); off = align256(off + E_ * 4);
  int*   offsets = (int*)(ws + off); off = align256(off + E_ * 4);
  int*   tE    = (int*)(ws + off);   off = align256(off + (size_t)MAXTILES * 4);
  int*   tS    = (int*)(ws + off);   off = align256(off + (size_t)MAXTILES * 4);
  int*   nT    = (int*)(ws + off);   off = align256(off + 4);

  long n4 = (long)N_TOK * D_ / 4;
  zero_init_kernel<<<(int)(n4 / 256), 256, 0, stream>>>(y, counts, n4);

  long n8 = (long)N_TOK * D_ / 8;
  cvt_bf16_kernel<<<(int)(n8 / 256), 256, 0, stream>>>(x, xb, n8);

  transpose_cvt_kernel<<<dim3(H_ / 32, D_ / 32, E_), dim3(32, 8), 0, stream>>>(W1, w1t, D_, H_);
  transpose_cvt_kernel<<<dim3(D_ / 32, H_ / 32, E_), dim3(32, 8), 0, stream>>>(W2, w2t, H_, D_);

  router_kernel<<<N_TOK / 8, 256, 0, stream>>>(x, Wg, tki, tkw, counts);
  plan_kernel<<<1, 1, 0, stream>>>(counts, offsets, tE, tS, nT);
  scatter_kernel<<<E_, 32, 0, stream>>>(tki, tkw, offsets, etok, ew);

  moe_ffn_kernel<<<MAXTILES, 256, 0, stream>>>(xb, w1t, w2t, b1, b2,
                                               etok, ew, counts, offsets,
                                               tE, tS, nT, y);
}
